// Encoder_47450798686673
// MI455X (gfx1250) — matrix operand order repaired and checked by reading, not compiled, not run on hardware
//
#include <hip/hip_runtime.h>
#include <hip/hip_bf16.h>

typedef __attribute__((ext_vector_type(8)))  __bf16 v8bf;
typedef __attribute__((ext_vector_type(16))) __bf16 v16bf;
typedef __attribute__((ext_vector_type(8)))  float  v8f;

#define C21   21
#define BGRP  2048
#define NNODE (C21*BGRP)        // 43008
#define EPER  420
#define HID   105
#define FIN   1024
#define KCH   5
#define N1    1280              // 5*256
#define C1    256
#define N2    320               // 5*64
#define C2    64

// workspace layout (float offsets; bf16 arrays use half-capacity)
#define WS_EW   0
#define WS_P    1024
#define WS_XHI  4096
#define WS_XLO  (WS_XHI + (NNODE*FIN)/2)
#define WS_W1HI (WS_XLO + (NNODE*FIN)/2)
#define WS_W1LO (WS_W1HI + (N1*FIN)/2)
#define WS_W2HI (WS_W1LO + (N1*FIN)/2)
#define WS_W2LO (WS_W2HI + (N2*C1)/2)
#define WS_Y    (WS_W2LO + (N2*C1)/2)
#define WS_HHI  (WS_Y + (size_t)NNODE*N1)
#define WS_HLO  (WS_HHI + (NNODE*C1)/2)
#define WS_Z    (WS_HLO + (NNODE*C1)/2)

// output layout: out(43008x64), ew(420), train_ew(860160)
#define OUT_EW  (NNODE*C2)
#define OUT_TEW (OUT_EW + EPER)

// issue one 16B-per-lane async global->LDS copy (ASYNCcnt tracked)
__device__ __forceinline__ void async_b128(unsigned lds_off, unsigned byte_off, const void* sbase)
{
    asm volatile("global_load_async_to_lds_b128 %0, %1, %2"
                 :: "v"(lds_off), "v"(byte_off), "s"(sbase) : "memory");
}
__device__ __forceinline__ void wait_async0()   // all async done
{
    asm volatile("s_wait_asynccnt 0" ::: "memory");
}
__device__ __forceinline__ void wait_async6()   // allow 6 newest (next chunk) in flight
{
    asm volatile("s_wait_asynccnt 6" ::: "memory");
}
__device__ __forceinline__ unsigned lds_off_of(const void* p)
{
    return (unsigned)(size_t)p;   // LDS aperture: addr[31:0] is the LDS byte offset
}

// ---------------------------------------------------------------------------
// Stage 1: edge-weight MLP, degree norm, A matrix, Chebyshev P_k (one block)
// ---------------------------------------------------------------------------
__global__ __launch_bounds__(512) void adj_kernel(
    const float* __restrict__ ew_in, const float* __restrict__ Wfc1,
    const float* __restrict__ Wfc2, float* __restrict__ ws, float* __restrict__ out)
{
    __shared__ float sEW[EPER], sH1[HID], sEW2[EPER], sDIS[C21];
    __shared__ float sAm[441], sT0[441], sT1[441];
    const int t = threadIdx.x;

    if (t < EPER) sEW[t] = ew_in[t];
    __syncthreads();

    if (t < HID) {                      // h1 = elu(ew^T @ Wfc1)
        float s = 0.f;
        for (int i = 0; i < EPER; ++i) s += sEW[i] * Wfc1[i*HID + t];
        sH1[t] = s > 0.f ? s : expm1f(s);
    }
    __syncthreads();

    if (t < EPER) {                     // ew' = relu(tanh(h1 @ Wfc2))
        float s = 0.f;
        for (int i = 0; i < HID; ++i) s += sH1[i] * Wfc2[i*EPER + t];
        float w = fmaxf(tanhf(s), 0.f);
        sEW2[t] = w;
        ws[WS_EW + t] = w;
        out[OUT_EW + t] = w;
    }
    __syncthreads();

    if (t < C21) {                      // deg / rsqrt (src-major edge order)
        float d = 0.f;
        for (int e = t*20; e < t*20 + 20; ++e) d += sEW2[e];
        sDIS[t] = d > 0.f ? rsqrtf(d) : 0.f;
    }
    if (t < 441) sAm[t] = 0.f;
    __syncthreads();

    if (t < EPER) {                     // A[dst][src] = -(dis[src]*w*dis[dst])
        int i = t / 20, jr = t % 20;
        int j = jr + (jr >= i ? 1 : 0);
        sAm[j*C21 + i] = -(sDIS[i] * sEW2[t] * sDIS[j]);
    }
    __syncthreads();

    if (t < 441) {                      // P0 = I, P1 = A
        sT0[t] = (t % 22 == 0) ? 1.f : 0.f;
        sT1[t] = sAm[t];
        ws[WS_P + t]       = sT0[t];
        ws[WS_P + 441 + t] = sT1[t];
    }
    __syncthreads();

    for (int s = 2; s < KCH; ++s) {     // P_s = 2 A P_{s-1} - P_{s-2}
        float val = 0.f;
        if (t < 441) {
            int r = t / C21, c = t % C21;
            float acc = 0.f;
            for (int i = 0; i < C21; ++i) acc += sAm[r*C21 + i] * sT1[i*C21 + c];
            val = 2.f*acc - sT0[t];
        }
        __syncthreads();
        if (t < 441) sT0[t] = sT1[t];
        __syncthreads();
        if (t < 441) { sT1[t] = val; ws[WS_P + s*441 + t] = val; }
        __syncthreads();
    }
}

// ---------------------------------------------------------------------------
// fp32 -> bf16 hi/lo split (x = hi + lo, |lo| ~ 2^-8 |x|)
// ---------------------------------------------------------------------------
__global__ void split_bf16(const float* __restrict__ X, __bf16* __restrict__ Hi,
                           __bf16* __restrict__ Lo, int total)
{
    for (int idx = blockIdx.x*blockDim.x + threadIdx.x; idx < total;
         idx += gridDim.x*blockDim.x) {
        float v = X[idx];
        __bf16 h = (__bf16)v;
        Hi[idx] = h;
        Lo[idx] = (__bf16)(v - (float)h);
    }
}

// Weight repack + split: W(k,f,o) -> K-major Wt(n = k*O+o, f), bf16 hi/lo
__global__ void pack_w1(const float* __restrict__ W1, __bf16* __restrict__ Hi,
                        __bf16* __restrict__ Lo)
{
    const int total = N1*FIN;
    for (int idx = blockIdx.x*blockDim.x + threadIdx.x; idx < total;
         idx += gridDim.x*blockDim.x) {
        int n = idx / FIN, f = idx % FIN;
        int k = n >> 8, o = n & 255;
        float w = W1[(k*FIN + f)*C1 + o];
        __bf16 h = (__bf16)w;
        Hi[idx] = h;
        Lo[idx] = (__bf16)(w - (float)h);
    }
}

__global__ void pack_w2(const float* __restrict__ W2, __bf16* __restrict__ Hi,
                        __bf16* __restrict__ Lo)
{
    const int total = N2*C1;
    for (int idx = blockIdx.x*blockDim.x + threadIdx.x; idx < total;
         idx += gridDim.x*blockDim.x) {
        int n = idx / C1, f = idx % C1;
        int k = n >> 6, o = n & 63;
        float w = W2[(k*C1 + f)*C2 + o];
        __bf16 h = (__bf16)w;
        Hi[idx] = h;
        Lo[idx] = (__bf16)(w - (float)h);
    }
}

// ---------------------------------------------------------------------------
// bf16x3 split GEMM: C(MxN,f32) = (Ahi+Alo)(MxK) @ (Bhi+Blo)(KxN), drop lo*lo.
// B operands pre-packed K-major (Bt[n][k]). Block tile 128x64, K-chunk 32.
// Double-buffered LDS, async tile fills (global_load_async_to_lds_b128);
// s_wait_asynccnt 6 keeps the next chunk's 6 copies in flight during compute.
// 8 waves, each a 32x32 subtile: 12x v_wmma_f32_16x16x32_bf16 per chunk/wave.
// ---------------------------------------------------------------------------
__global__ __launch_bounds__(256) void gemm_bf16x3(
    const __bf16* __restrict__ Ahi, const __bf16* __restrict__ Alo,
    const __bf16* __restrict__ BhiT, const __bf16* __restrict__ BloT,
    float* __restrict__ C, int M, int N, int K)
{
    __shared__ __bf16 sAhi[2][128*32];   // 2 x 8KB
    __shared__ __bf16 sAlo[2][128*32];   // 2 x 8KB
    __shared__ __bf16 sBhi[2][64*32];    // 2 x 4KB (K-major rows)
    __shared__ __bf16 sBlo[2][64*32];    // 2 x 4KB
    const int tid  = threadIdx.x;
    const int lane = tid & 31;
    const int wave = tid >> 5;
    const int wm   = (wave >> 1) * 32;
    const int wn   = (wave & 1)  * 32;
    const int mblk = blockIdx.y * 128;
    const int nblk = blockIdx.x * 64;
    const int l15   = lane & 15;
    const int koffA = (lane >> 4) << 3;   // 0 / 8   (bf16 elems)
    const int koffB = (lane >> 4) << 3;   // 0 / 8

    // per-thread 16B segment coordinates for tile fills
    const int ra0 = tid >> 2,            sa0 = (tid & 3) << 3;        // A segs 0..255
    const int ra1 = (tid + 256) >> 2,    sa1 = ((tid + 256) & 3) << 3;// A segs 256..511
    const int rb  = tid >> 2,            sb  = (tid & 3) << 3;        // B segs 0..255

    unsigned ldsAhi[2] = { lds_off_of(&sAhi[0][0]), lds_off_of(&sAhi[1][0]) };
    unsigned ldsAlo[2] = { lds_off_of(&sAlo[0][0]), lds_off_of(&sAlo[1][0]) };
    unsigned ldsBhi[2] = { lds_off_of(&sBhi[0][0]), lds_off_of(&sBhi[1][0]) };
    unsigned ldsBlo[2] = { lds_off_of(&sBlo[0][0]), lds_off_of(&sBlo[1][0]) };

    // issue the 6 async copies for one K-chunk into buffer `buf`
    auto issue = [&](int kc, int buf) {
        unsigned gA0 = (unsigned)(((size_t)(mblk + ra0)*K + kc + sa0) * 2);
        unsigned gA1 = (unsigned)(((size_t)(mblk + ra1)*K + kc + sa1) * 2);
        unsigned gB  = (unsigned)(((size_t)(nblk + rb )*K + kc + sb ) * 2);
        async_b128(ldsAhi[buf] + (ra0*32 + sa0)*2, gA0, Ahi);
        async_b128(ldsAhi[buf] + (ra1*32 + sa1)*2, gA1, Ahi);
        async_b128(ldsAlo[buf] + (ra0*32 + sa0)*2, gA0, Alo);
        async_b128(ldsAlo[buf] + (ra1*32 + sa1)*2, gA1, Alo);
        async_b128(ldsBhi[buf] + (rb*32 + sb)*2,  gB,  BhiT);
        async_b128(ldsBlo[buf] + (rb*32 + sb)*2,  gB,  BloT);
    };

    v8f acc[2][2] = {};
    const int nch = K >> 5;

    issue(0, 0);
    for (int c = 0; c < nch; ++c) {
        const int cur = c & 1;
        if (c + 1 < nch) { issue((c + 1) << 5, cur ^ 1); wait_async6(); }
        else             { wait_async0(); }
        __syncthreads();

        v16bf ahi[2], alo[2], bhi[2], blo[2];
        #pragma unroll
        for (int i = 0; i < 2; ++i) {
            const __bf16* pa = &sAhi[cur][(wm + i*16 + l15)*32];
            const __bf16* pl = &sAlo[cur][(wm + i*16 + l15)*32];
            v8bf h0 = *(const v8bf*)(pa + koffA);
            v8bf h1 = *(const v8bf*)(pa + koffA + 16);
            v8bf l0 = *(const v8bf*)(pl + koffA);
            v8bf l1 = *(const v8bf*)(pl + koffA + 16);
            ahi[i] = __builtin_shufflevector(h0, h1, 0,1,2,3,4,5,6,7,8,9,10,11,12,13,14,15);
            alo[i] = __builtin_shufflevector(l0, l1, 0,1,2,3,4,5,6,7,8,9,10,11,12,13,14,15);
        }
        #pragma unroll
        for (int j = 0; j < 2; ++j) {
            const __bf16* pb = &sBhi[cur][(wn + j*16 + l15)*32];
            const __bf16* pl = &sBlo[cur][(wn + j*16 + l15)*32];
            v8bf h0 = *(const v8bf*)(pb + koffB);
            v8bf h1 = *(const v8bf*)(pb + koffB + 16);
            v8bf l0 = *(const v8bf*)(pl + koffB);
            v8bf l1 = *(const v8bf*)(pl + koffB + 16);
            bhi[j] = __builtin_shufflevector(h0, h1, 0,1,2,3,4,5,6,7,8,9,10,11,12,13,14,15);
            blo[j] = __builtin_shufflevector(l0, l1, 0,1,2,3,4,5,6,7,8,9,10,11,12,13,14,15);
        }

        #pragma unroll
        for (int i = 0; i < 2; ++i)
            #pragma unroll
            for (int j = 0; j < 2; ++j) {
                acc[i][j] = __builtin_amdgcn_wmma_f32_16x16x32_bf16(
                    false, ahi[i], false, bhi[j], (short)0, acc[i][j], false, false);
                acc[i][j] = __builtin_amdgcn_wmma_f32_16x16x32_bf16(
                    false, ahi[i], false, blo[j], (short)0, acc[i][j], false, false);
                acc[i][j] = __builtin_amdgcn_wmma_f32_16x16x32_bf16(
                    false, alo[i], false, bhi[j], (short)0, acc[i][j], false, false);
            }
        __syncthreads();   // buffer cur free for reuse two iterations later
    }

    // C layout: VGPR v -> M = v (lanes 0-15) / v+8 (lanes 16-31); N = lane&15
    #pragma unroll
    for (int i = 0; i < 2; ++i)
        #pragma unroll
        for (int j = 0; j < 2; ++j) {
            int rbase = mblk + wm + i*16 + ((lane < 16) ? 0 : 8);
            int col   = nblk + wn + j*16 + l15;
            #pragma unroll
            for (int v = 0; v < 8; ++v)
                C[(size_t)(rbase + v)*N + col] = acc[i][j][v];
        }
}

// ---------------------------------------------------------------------------
// H = relu(sum_k P_k (21x21) @ Y[:,k*256:] + b1), emitted as bf16 hi/lo split
// ---------------------------------------------------------------------------
__global__ __launch_bounds__(256) void prop1_relu(
    const float* __restrict__ Y, const float* __restrict__ P,
    const float* __restrict__ b1, __bf16* __restrict__ Hhi, __bf16* __restrict__ Hlo)
{
    extern __shared__ float sm[];
    float* sY = sm;                 // 21*1280
    float* sP = sm + C21*N1;        // 5*441
    const int g = blockIdx.x, t = threadIdx.x;
    for (int idx = t; idx < C21*N1; idx += 256) sY[idx] = Y[(size_t)g*(C21*N1) + idx];
    for (int idx = t; idx < KCH*441; idx += 256) sP[idx] = P[idx];
    __syncthreads();
    const float bias = b1[t];
    for (int r = 0; r < C21; ++r) {
        float acc = bias;
        #pragma unroll
        for (int k = 0; k < KCH; ++k) {
            const float* Pk = &sP[k*441 + r*C21];
            const float* Yk = &sY[k*C1 + t];
            #pragma unroll
            for (int i = 0; i < C21; ++i) acc += Pk[i] * Yk[i*N1];
        }
        float h = fmaxf(acc, 0.f);
        __bf16 hh = (__bf16)h;
        size_t o = (size_t)(g*C21 + r)*C1 + t;
        Hhi[o] = hh;
        Hlo[o] = (__bf16)(h - (float)hh);
    }
}

// out = sum_k P_k @ Z[:,k*64:] + b2   (no relu), fp32 to d_out
__global__ __launch_bounds__(64) void prop2_bias(
    const float* __restrict__ Z, const float* __restrict__ P,
    const float* __restrict__ b2, float* __restrict__ Out)
{
    __shared__ float sZ[C21*N2];
    __shared__ float sP[KCH*441];
    const int g = blockIdx.x, t = threadIdx.x;
    for (int idx = t; idx < C21*N2; idx += 64) sZ[idx] = Z[(size_t)g*(C21*N2) + idx];
    for (int idx = t; idx < KCH*441; idx += 64) sP[idx] = P[idx];
    __syncthreads();
    const float bias = b2[t];
    for (int r = 0; r < C21; ++r) {
        float acc = bias;
        #pragma unroll
        for (int k = 0; k < KCH; ++k)
            #pragma unroll
            for (int i = 0; i < C21; ++i)
                acc += sP[k*441 + r*C21 + i] * sZ[i*N2 + k*C2 + t];
        Out[(size_t)(g*C21 + r)*C2 + t] = acc;
    }
}

__global__ void tile_ew(const float* __restrict__ ws, float* __restrict__ out)
{
    const int total = EPER*BGRP;
    for (int idx = blockIdx.x*blockDim.x + threadIdx.x; idx < total;
         idx += gridDim.x*blockDim.x)
        out[OUT_TEW + idx] = ws[WS_EW + idx % EPER];
}

// ---------------------------------------------------------------------------
extern "C" void kernel_launch(void* const* d_in, const int* in_sizes, int n_in,
                              void* d_out, int out_size, void* d_ws, size_t ws_size,
                              hipStream_t stream)
{
    const float* x    = (const float*)d_in[0];
    // d_in[1] = edge_index (unused: fixed block-diagonal structure exploited)
    const float* ew   = (const float*)d_in[2];
    const float* Wfc1 = (const float*)d_in[3];
    const float* Wfc2 = (const float*)d_in[4];
    const float* W1   = (const float*)d_in[5];
    const float* b1   = (const float*)d_in[6];
    const float* W2   = (const float*)d_in[7];
    const float* b2   = (const float*)d_in[8];
    float* out = (float*)d_out;
    float* ws  = (float*)d_ws;

    __bf16* xhi  = (__bf16*)(ws + WS_XHI);
    __bf16* xlo  = (__bf16*)(ws + WS_XLO);
    __bf16* w1hi = (__bf16*)(ws + WS_W1HI);
    __bf16* w1lo = (__bf16*)(ws + WS_W1LO);
    __bf16* w2hi = (__bf16*)(ws + WS_W2HI);
    __bf16* w2lo = (__bf16*)(ws + WS_W2LO);
    __bf16* hhi  = (__bf16*)(ws + WS_HHI);
    __bf16* hlo  = (__bf16*)(ws + WS_HLO);
    float*  Y    = ws + WS_Y;
    float*  Z    = ws + WS_Z;

    adj_kernel<<<1, 512, 0, stream>>>(ew, Wfc1, Wfc2, ws, out);
    split_bf16<<<4096, 256, 0, stream>>>(x, xhi, xlo, NNODE*FIN);
    pack_w1<<<1280, 256, 0, stream>>>(W1, w1hi, w1lo);
    pack_w2<<<80, 256, 0, stream>>>(W2, w2hi, w2lo);

    // GEMM1: Y = X @ Wcat1   (43008 x 1024 x 1280)
    gemm_bf16x3<<<dim3(N1/64, NNODE/128), 256, 0, stream>>>(
        xhi, xlo, w1hi, w1lo, Y, NNODE, N1, FIN);

    size_t sh1 = (size_t)(C21*N1 + KCH*441) * sizeof(float);
    prop1_relu<<<BGRP, 256, sh1, stream>>>(Y, ws + WS_P, b1, hhi, hlo);

    // GEMM2: Z = H @ Wcat2   (43008 x 256 x 320)
    gemm_bf16x3<<<dim3(N2/64, NNODE/128), 256, 0, stream>>>(
        hhi, hlo, w2hi, w2lo, Z, NNODE, N2, C1);

    prop2_bias<<<BGRP, 64, 0, stream>>>(Z, ws + WS_P, b2, out);
    tile_ew<<<840, 256, 0, stream>>>(ws, out);
}
